// Up_SSD_11811160064576
// MI455X (gfx1250) — matrix operand order repaired and checked by reading, not compiled, not run on hardware
//
#include <hip/hip_runtime.h>
#include <hip/hip_bf16.h>
#include <math.h>

// ---------------- problem constants (match reference) ----------------
#define B_      4
#define IN_DIM  128
#define L_      2048
#define D_MODEL 256
#define OUT_DIM 128
#define D_STATE 16
#define K_SAME  4
#define HEADDIM 3
#define D_CONV  4
#define D_INNER 768          // EXPAND * D_MODEL
#define NHEADS  256          // D_INNER / HEADDIM
#define CONV_DIM 800         // D_INNER + 2*D_STATE
#define D_IN_PROJ 1824       // 2*D_INNER + 2*D_STATE + NHEADS
#define MROWS   (B_ * L_)    // 8192
#define EPSV    1e-5f
#define K_CIN   (IN_DIM * K_SAME)    // 512  (im2col K for input conv)
#define K_COUT  (D_MODEL * K_SAME)   // 1024 (im2col K for output conv)

typedef _Float16 h8_t  __attribute__((ext_vector_type(8)));
typedef _Float16 v16h  __attribute__((ext_vector_type(16)));
typedef float    v8f   __attribute__((ext_vector_type(8)));

__device__ __forceinline__ float silu_f(float x) { return x / (1.f + __expf(-x)); }
__device__ __forceinline__ float softplus_f(float x) {
    return (x > 20.f) ? x : log1pf(__expf(x));
}

// =====================================================================
// Generic WMMA GEMM: C[M,N] = A16[M,K] * B16[K,N] (+ bias[n]), f32 accum
// block = 128 threads (4 waves). Block tile: 64(M) x 32(N).
// Each wave owns one 16-row M tile; the 32-wide N panel of B is staged
// in LDS (transposed to the WMMA B packing) and shared by all 4 waves.
// Per K-step per wave: 2x global b128 (A) + 4x ds_load_b128 (B) + 2 WMMA.
// =====================================================================
template<bool F16OUT>
__global__ void k_gemm_wmma(const _Float16* __restrict__ A, const _Float16* __restrict__ Bw,
                            const float* __restrict__ bias,
                            float* __restrict__ Cf, _Float16* __restrict__ Ch,
                            int Kdim, int Ndim) {
    constexpr int LDB = 40;                 // LDS row stride (halves): 80B, 16B-aligned,
                                            // conflict-free across 16 lanes of a subtile
    __shared__ _Float16 ldsB[32 * LDB];     // 2560 B
    const int t    = threadIdx.x;
    const int wave = t >> 5;
    const int lane = t & 31;
    const int kh   = lane >> 4;             // K-half selector
    const int ml   = lane & 15;             // row (A) / col (B,C,D)
    const int n0   = blockIdx.x * 32;
    const int m0   = (blockIdx.y * 4 + wave) * 16;

    const _Float16* Arow = A + (size_t)(m0 + ml) * Kdim;
    v8f acc0 = {}, acc1 = {};

    for (int kb = 0; kb < Kdim; kb += 32) {
        // ---- stage B[kb..kb+31, n0..n0+31] into LDS, transposed: ldsB[n*LDB + k]
        #pragma unroll
        for (int j = 0; j < 4; ++j) {
            int idx = t + j * 128;          // 512 dword slots: 32 K-rows x 16 dwords
            int k   = idx >> 4;
            int nd  = idx & 15;
            const _Float16* gp = Bw + (size_t)(kb + k) * Ndim + n0 + nd * 2;
            _Float16 b0 = gp[0], b1 = gp[1];            // one coalesced b32 load
            ldsB[(nd * 2 + 0) * LDB + k] = b0;
            ldsB[(nd * 2 + 1) * LDB + k] = b1;
        }
        __syncthreads();

        // ---- A operand: lane (m, kh) holds K = kb+8kh..+7 and kb+16+8kh..+7
        h8_t a0 = *(const h8_t*)(Arow + kb + 8 * kh);
        h8_t a1 = *(const h8_t*)(Arow + kb + 16 + 8 * kh);
        v16h a;
        #pragma unroll
        for (int i = 0; i < 8; ++i) { a[i] = a0[i]; a[i + 8] = a1[i]; }

        if (kb + 32 < Kdim) __builtin_prefetch(Arow + kb + 32, 0, 3);  // global_prefetch_b8

        // ---- B operands from LDS: subtile s, lane col n = s*16+ml, K = 8kh+e and 16+8kh+e
        const _Float16* l0 = &ldsB[(0 * 16 + ml) * LDB + 8 * kh];
        const _Float16* l1 = &ldsB[(1 * 16 + ml) * LDB + 8 * kh];
        h8_t p0 = *(const h8_t*)l0, p1 = *(const h8_t*)(l0 + 16);
        h8_t q0 = *(const h8_t*)l1, q1 = *(const h8_t*)(l1 + 16);
        v16h bv0, bv1;
        #pragma unroll
        for (int i = 0; i < 8; ++i) {
            bv0[i] = p0[i]; bv0[i + 8] = p1[i];
            bv1[i] = q0[i]; bv1[i + 8] = q1[i];
        }

        acc0 = __builtin_amdgcn_wmma_f32_16x16x32_f16(false, a, false, bv0, (short)0, acc0, false, false);
        acc1 = __builtin_amdgcn_wmma_f32_16x16x32_f16(false, a, false, bv1, (short)0, acc1, false, false);
        __syncthreads();
    }

    // ---- epilogue: D layout vgpr r -> row m = r + 8kh, col n = ml
    const int col = n0 + ml;
    float bv0 = bias ? bias[col]      : 0.f;
    float bv1 = bias ? bias[col + 16] : 0.f;
    if constexpr (F16OUT) {
        _Float16* Cp = Ch + (size_t)m0 * Ndim + col;
        #pragma unroll
        for (int r = 0; r < 8; ++r) {
            size_t row = (size_t)(r + 8 * kh) * Ndim;
            Cp[row]      = (_Float16)(acc0[r] + bv0);
            Cp[row + 16] = (_Float16)(acc1[r] + bv1);
        }
    } else {
        float* Cp = Cf + (size_t)m0 * Ndim + col;
        #pragma unroll
        for (int r = 0; r < 8; ++r) {
            size_t row = (size_t)(r + 8 * kh) * Ndim;
            Cp[row]      = acc0[r] + bv0;
            Cp[row + 16] = acc1[r] + bv1;
        }
    }
}

// ---------------- im2col of x: A1[(b,l), i*4+k] = x[b,i,l-1+k] ----------
__global__ void k_im2col_x(const float* __restrict__ x, _Float16* __restrict__ A1) {
    int idx = blockIdx.x * 256 + threadIdx.x;       // MROWS * 512
    if (idx >= MROWS * K_CIN) return;
    int m = idx >> 9, c = idx & 511;
    int i = c >> 2, k = c & 3;
    int b = m / L_, l = m % L_;
    int li = l - 1 + k;
    float v = (li >= 0 && li < L_) ? x[((size_t)(b * IN_DIM + i)) * L_ + li] : 0.f;
    A1[idx] = (_Float16)v;
}

// ---------------- conv_w [O,I,K] -> W1c [I*K, O] f16 --------------------
__global__ void k_wt_conv_in(const float* __restrict__ cw, _Float16* __restrict__ w) {
    int idx = blockIdx.x * 256 + threadIdx.x;       // 512 * 256
    if (idx >= K_CIN * D_MODEL) return;
    int r = idx / D_MODEL, o = idx % D_MODEL;
    w[idx] = (_Float16)cw[(size_t)o * K_CIN + r];
}

// ---------------- f32 -> f16 conversion (weights) -----------------------
__global__ void k_cvt_f16(const float* __restrict__ s, _Float16* __restrict__ d, int n) {
    int i = blockIdx.x * 256 + threadIdx.x;
    if (i < n) d[i] = (_Float16)s[i];
}

// ---------------- dt = softplus(dt_raw + dt_bias) -----------------------
__global__ void k_dt(const float* __restrict__ zx, const float* __restrict__ dt_bias,
                     float* __restrict__ dtv) {
    int idx = blockIdx.x * 256 + threadIdx.x;       // MROWS*NHEADS
    if (idx >= MROWS * NHEADS) return;
    int m = idx / NHEADS, h = idx % NHEADS;
    float v = zx[(size_t)m * D_IN_PROJ + (D_INNER + CONV_DIM) + h] + dt_bias[h];
    dtv[idx] = softplus_f(v);
}

// ---------------- causal depthwise conv + SiLU over xBC -----------------
__global__ void k_dwconv(const float* __restrict__ zx, const float* __restrict__ dww,
                         const float* __restrict__ dwb, float* __restrict__ xact) {
    int row = blockIdx.x;                 // b*L + l
    int b = row / L_, l = row % L_;
    for (int c = threadIdx.x; c < CONV_DIM; c += 256) {
        float acc = dwb[c];
        #pragma unroll
        for (int k = 0; k < D_CONV; ++k) {
            int li = l - (D_CONV - 1) + k;             // causal pad (3,0)
            if (li >= 0)
                acc += zx[((size_t)(b * L_ + li)) * D_IN_PROJ + D_INNER + c] * dww[c * D_CONV + k];
        }
        xact[(size_t)row * CONV_DIM + c] = silu_f(acc);
    }
}

// ---------------- SSD scan: 16-lane group per (b, head, p) --------------
__global__ void k_ssd(const float* __restrict__ dtv, const float* __restrict__ xact,
                      const float* __restrict__ A_log, const float* __restrict__ Dp,
                      float* __restrict__ ybuf) {
    int g = blockIdx.x * 16 + (threadIdx.x >> 4);    // 3072 groups
    int n = threadIdx.x & 15;                        // state index
    int b    = g / (NHEADS * HEADDIM);
    int rem  = g % (NHEADS * HEADDIM);
    int head = rem / HEADDIM;
    int p    = rem % HEADDIM;

    float Ah = -__expf(A_log[head]);
    float Dh = Dp[head];
    float h  = 0.f;

    for (int l = 0; l < L_; ++l) {
        size_t row = (size_t)b * L_ + l;
        float dt = dtv[row * NHEADS + head];
        float dA = __expf(dt * Ah);
        float xv = xact[row * CONV_DIM + head * HEADDIM + p];
        float Bv = xact[row * CONV_DIM + D_INNER + n];
        float Cv = xact[row * CONV_DIM + D_INNER + D_STATE + n];
        h = h * dA + dt * xv * Bv;
        float s = h * Cv;
        s += __shfl_xor(s, 1);
        s += __shfl_xor(s, 2);
        s += __shfl_xor(s, 4);
        s += __shfl_xor(s, 8);
        if (n == 0) ybuf[row * D_INNER + head * HEADDIM + p] = s + Dh * xv;
    }
}

// ---------------- gate with silu(z), RMS-norm, cast to f16 --------------
__global__ void k_gatenorm(const float* __restrict__ ybuf, const float* __restrict__ zx,
                           const float* __restrict__ norm_w, _Float16* __restrict__ y16) {
    int row = blockIdx.x;
    int t = threadIdx.x;
    const float* z = zx + (size_t)row * D_IN_PROJ;
    const float* y = ybuf + (size_t)row * D_INNER;
    float g[3], ss = 0.f;
    #pragma unroll
    for (int j = 0; j < 3; ++j) {
        int d = t * 3 + j;
        float gv = y[d] * silu_f(z[d]);
        g[j] = gv;
        ss += gv * gv;
    }
    __shared__ float red[256];
    red[t] = ss;
    __syncthreads();
    for (int s2 = 128; s2 > 0; s2 >>= 1) {
        if (t < s2) red[t] += red[t + s2];
        __syncthreads();
    }
    float rms = rsqrtf(red[0] / (float)D_INNER + EPSV);
    #pragma unroll
    for (int j = 0; j < 3; ++j) {
        int d = t * 3 + j;
        y16[(size_t)row * D_INNER + d] = (_Float16)(g[j] * rms * norm_w[d]);
    }
}

// ---------------- im2col of om16: A2[(b,l), k*256+d] = om16[b,l-1+k,d] --
__global__ void k_im2col_om(const _Float16* __restrict__ om16, _Float16* __restrict__ A2) {
    int idx = blockIdx.x * 256 + threadIdx.x;       // MROWS * 1024
    if (idx >= MROWS * K_COUT) return;
    int m = idx >> 10, c = idx & 1023;
    int k = c >> 8, d = c & 255;
    int b = m / L_, l = m % L_;
    int li = l - 1 + k;
    A2[idx] = (li >= 0 && li < L_) ? om16[((size_t)(b * L_ + li)) * D_MODEL + d]
                                   : (_Float16)0.f;
}

// ---------------- out_conv_w [O,D,K] -> W2c [K*256+D, O] f16 ------------
__global__ void k_wt_conv_out(const float* __restrict__ cw, _Float16* __restrict__ w) {
    int idx = blockIdx.x * 256 + threadIdx.x;       // 1024 * 128
    if (idx >= K_COUT * OUT_DIM) return;
    int r = idx / OUT_DIM, o = idx % OUT_DIM;
    int k = r >> 8, d = r & 255;
    w[idx] = (_Float16)cw[(size_t)o * K_COUT + d * K_SAME + k];
}

// ---------------- [B*L, 128] -> [B, 128, L] transpose -------------------
__global__ void k_transpose_out(const float* __restrict__ tmpO, float* __restrict__ out) {
    int idx = blockIdx.x * 256 + threadIdx.x;       // B_*OUT_DIM*L_
    if (idx >= B_ * OUT_DIM * L_) return;
    int b = idx / (OUT_DIM * L_);
    int rem = idx % (OUT_DIM * L_);
    int o = rem / L_, l = rem % L_;
    out[idx] = tmpO[((size_t)(b * L_ + l)) * OUT_DIM + o];
}

extern "C" void kernel_launch(void* const* d_in, const int* in_sizes, int n_in,
                              void* d_out, int out_size, void* d_ws, size_t ws_size,
                              hipStream_t stream) {
    (void)in_sizes; (void)n_in; (void)out_size; (void)ws_size;
    const float* x          = (const float*)d_in[0];
    const float* conv_w     = (const float*)d_in[1];
    const float* conv_b     = (const float*)d_in[2];
    const float* in_proj_w  = (const float*)d_in[3];
    const float* dw_conv_w  = (const float*)d_in[4];
    const float* dw_conv_b  = (const float*)d_in[5];
    const float* dt_bias    = (const float*)d_in[6];
    const float* A_log      = (const float*)d_in[7];
    const float* Dp         = (const float*)d_in[8];
    const float* norm_w     = (const float*)d_in[9];
    const float* mamba_out_w= (const float*)d_in[10];
    const float* out_conv_w = (const float*)d_in[11];
    float* out = (float*)d_out;

    // ---- workspace carve-out (256B aligned) ----
    char* p = (char*)d_ws;
    auto alloc = [&](size_t bytes) { char* r = p; p += (bytes + 255) & ~(size_t)255; return r; };
    _Float16* A1   = (_Float16*)alloc((size_t)MROWS * K_CIN * 2);       // 8 MB im2col(x)
    _Float16* W1c  = (_Float16*)alloc((size_t)K_CIN * D_MODEL * 2);     // conv_in weights
    _Float16* u16  = (_Float16*)alloc((size_t)MROWS * D_MODEL * 2);     // conv_in output
    _Float16* W1h  = (_Float16*)alloc((size_t)D_MODEL * D_IN_PROJ * 2); // in_proj f16
    float*    zx   = (float*)alloc((size_t)MROWS * D_IN_PROJ * 4);      // z | xBC | dt
    float*    xact = (float*)alloc((size_t)MROWS * CONV_DIM * 4);       // silu(dwconv(xBC))
    float*    dtv  = (float*)alloc((size_t)MROWS * NHEADS * 4);         // softplus dt
    float*    ybuf = (float*)alloc((size_t)MROWS * D_INNER * 4);        // ssd output
    _Float16* y16  = (_Float16*)alloc((size_t)MROWS * D_INNER * 2);     // gated+normed
    _Float16* W2h  = (_Float16*)alloc((size_t)D_INNER * D_MODEL * 2);   // mamba_out f16
    _Float16* om16 = (_Float16*)alloc((size_t)MROWS * D_MODEL * 2);     // out-proj result
    _Float16* A2   = (_Float16*)alloc((size_t)MROWS * K_COUT * 2);      // 16 MB im2col(om)
    _Float16* W2c  = (_Float16*)alloc((size_t)K_COUT * OUT_DIM * 2);    // conv_out weights
    float*    tmpO = (float*)alloc((size_t)MROWS * OUT_DIM * 4);        // pre-transpose

    // ---- weight prep + im2col(x) ----
    k_im2col_x<<<dim3(MROWS * K_CIN / 256), dim3(256), 0, stream>>>(x, A1);
    k_wt_conv_in<<<dim3(K_CIN * D_MODEL / 256), dim3(256), 0, stream>>>(conv_w, W1c);
    k_cvt_f16<<<dim3((D_MODEL * D_IN_PROJ + 255) / 256), dim3(256), 0, stream>>>(
        in_proj_w, W1h, D_MODEL * D_IN_PROJ);
    k_cvt_f16<<<dim3((D_INNER * D_MODEL + 255) / 256), dim3(256), 0, stream>>>(
        mamba_out_w, W2h, D_INNER * D_MODEL);
    k_wt_conv_out<<<dim3(K_COUT * OUT_DIM / 256), dim3(256), 0, stream>>>(out_conv_w, W2c);

    // ---- conv_in as WMMA GEMM: [8192,512] x [512,256] + conv_b -> u16 (f16)
    k_gemm_wmma<true><<<dim3(D_MODEL / 32, MROWS / 64), dim3(128), 0, stream>>>(
        A1, W1c, conv_b, nullptr, u16, K_CIN, D_MODEL);

    // ---- in-projection: [8192,256] x [256,1824] -> zx (f32)
    k_gemm_wmma<false><<<dim3(D_IN_PROJ / 32, MROWS / 64), dim3(128), 0, stream>>>(
        u16, W1h, nullptr, zx, nullptr, D_MODEL, D_IN_PROJ);

    // ---- pointwise + scan chain ----
    k_dt<<<dim3(MROWS * NHEADS / 256), dim3(256), 0, stream>>>(zx, dt_bias, dtv);
    k_dwconv<<<dim3(MROWS), dim3(256), 0, stream>>>(zx, dw_conv_w, dw_conv_b, xact);
    k_ssd<<<dim3(B_ * NHEADS * HEADDIM / 16), dim3(256), 0, stream>>>(
        dtv, xact, A_log, Dp, ybuf);
    k_gatenorm<<<dim3(MROWS), dim3(256), 0, stream>>>(ybuf, zx, norm_w, y16);

    // ---- out-projection: [8192,768] x [768,256] -> om16 (f16)
    k_gemm_wmma<true><<<dim3(D_MODEL / 32, MROWS / 64), dim3(128), 0, stream>>>(
        y16, W2h, nullptr, nullptr, om16, D_INNER, D_MODEL);

    // ---- conv_out as WMMA GEMM: im2col(om16) [8192,1024] x [1024,128] -> tmpO
    k_im2col_om<<<dim3(MROWS * K_COUT / 256), dim3(256), 0, stream>>>(om16, A2);
    k_gemm_wmma<false><<<dim3(OUT_DIM / 32, MROWS / 64), dim3(128), 0, stream>>>(
        A2, W2c, nullptr, tmpO, nullptr, K_COUT, OUT_DIM);

    // ---- final transpose to [B, 128, L] ----
    k_transpose_out<<<dim3(B_ * OUT_DIM * L_ / 256), dim3(256), 0, stream>>>(tmpO, out);
}
